// StaticGraphLSTMCell__90958817395182
// MI455X (gfx1250) — matrix operand order repaired and checked by reading, not compiled, not run on hardware
//
#include <hip/hip_runtime.h>

// ---------------------------------------------------------------------------
// GraphLSTM cell for MI455X (gfx1250, wave32, WMMA, async-to-LDS).
// B=4096 batch, N=24 nodes, I=64 in, H=256 hidden, G4=4H=1024.
//
// Pipeline:
//   k_gx        : normalize G -> gx (bf16, zero-padded 32x32) + gx_out (f32)
//   k_tobf x4   : one-time f32 -> bf16 conversion of x, hx, Wih, Whh
//   k_node_gemm : s[n][b][o] (bf16) = x[b,n,:]@Wih[n].T + hx[b,n,:]@Whh[n].T
//                 double-buffered LDS tiles via global_load_async_to_lds_b128,
//                 v_wmma_f32_16x16x32_bf16 compute, coalesced [n][b][o] stores
//   k_mix_lstm  : LDS-transpose s slice, gates[b,m,o] = sum_n gx[m,n]*s[n,b,o]
//                 (one padded K=32 WMMA) + biases + LSTM + clockwork mask
// ---------------------------------------------------------------------------

typedef __bf16 v16bf __attribute__((ext_vector_type(16)));
typedef __bf16 v8bf  __attribute__((ext_vector_type(8)));
typedef float  v8f   __attribute__((ext_vector_type(8)));
typedef float  v4f   __attribute__((ext_vector_type(4)));

#define B_  4096
#define N_  24
#define I_  64
#define H_  256
#define G4_ 1024
#define LROW_ 40         // LDS row stride in bf16 elements (80B, bank-skewed)

static __device__ __forceinline__ v4f ld4(const float* p) {
    return *reinterpret_cast<const v4f*>(p);
}
static __device__ __forceinline__ float sigmoidf(float v) {
    return 1.0f / (1.0f + __expf(-v));
}

// CDNA5 async copy: 16B per lane, global -> LDS, tracked by ASYNCcnt.
static __device__ __forceinline__ void async_ld16(unsigned lds_off, const void* g) {
    asm volatile("global_load_async_to_lds_b128 %0, %1, off"
                 :: "v"(lds_off), "v"(g) : "memory");
}
static __device__ __forceinline__ void wait_async6() {
    asm volatile("s_wait_asynccnt 6" ::: "memory");
}
static __device__ __forceinline__ void wait_async0() {
    asm volatile("s_wait_asynccnt 0" ::: "memory");
}

// ---------------------------------------------------------------------------
// Kernel 0: column-normalize G.  gx = G / colsum(G); gx_out = gx / colsum(gx).
// Pads (rows/cols 24..31) MUST be zero: they are the K/M pad slots of the
// stage-2 WMMA A operand.
// ---------------------------------------------------------------------------
__global__ __launch_bounds__(256)
void k_gx(const float* __restrict__ G, __bf16* __restrict__ gxb,
          float* __restrict__ gx_out)
{
    for (int idx = threadIdx.x; idx < 32 * 32; idx += 256) {
        const int m = idx >> 5, nn = idx & 31;
        __bf16 v = (__bf16)0.0f;
        if (m < N_ && nn < N_) {
            float cs = 0.0f;
            for (int mm = 0; mm < N_; ++mm) cs += G[mm * N_ + nn];
            const float gxv = G[m * N_ + nn] / cs;
            float cs2 = 0.0f;
            for (int mm = 0; mm < N_; ++mm) cs2 += G[mm * N_ + nn] / cs;
            gx_out[m * N_ + nn] = gxv / cs2;
            v = (__bf16)gxv;
        }
        gxb[idx] = v;
    }
}

// ---------------------------------------------------------------------------
// One-time f32 -> bf16 conversion (8 elements / thread, b128 in, b128 out).
// ---------------------------------------------------------------------------
__global__ __launch_bounds__(256)
void k_tobf(const float* __restrict__ src, __bf16* __restrict__ dst, int n8)
{
    const int i = blockIdx.x * 256 + threadIdx.x;
    if (i < n8) {
        const v4f a = ld4(src + (size_t)i * 8);
        const v4f b = ld4(src + (size_t)i * 8 + 4);
        v8bf o;
        #pragma unroll
        for (int e = 0; e < 4; ++e) { o[e] = (__bf16)a[e]; o[4 + e] = (__bf16)b[e]; }
        *reinterpret_cast<v8bf*>(dst + (size_t)i * 8) = o;
    }
}

// ---------------------------------------------------------------------------
// Kernel 1: per-node GEMMs with async-LDS double buffering.
// Grid: (n=24, b/128=32, o/256=4), 256 threads = 8 waves as 4(b) x 2(o).
// Each wave: 32(b) x 128(o) tile = 2x8 v8f accumulators.
// 10 K-steps of 32 (2 over I with Wih, 8 over H with Whh). Per step the WG
// async-copies a 256x32 weight tile (16KB) and 128x32 activation tile (8KB)
// into LDS (6 x b128 per thread), then 16 WMMAs per wave from ds_load_b128.
// Operand layouts (ISA 7.12.2):
//   A: lane<16: M=lane,    K={ks+0..7, ks+16..23}; lane>=16: M=lane-16, +8
//   B: lane<16: N=lane, K={ks+0..7, ks+16..23};    lane>=16: N=lane-16, +8
// Output stored [n][b][o]: 32B-contiguous runs per half-wave per vgpr row.
// ---------------------------------------------------------------------------
__global__ __launch_bounds__(256)
void k_node_gemm(const __bf16* __restrict__ xb, const __bf16* __restrict__ hb,
                 const __bf16* __restrict__ Wihb, const __bf16* __restrict__ Whhb,
                 __bf16* __restrict__ s)
{
    __shared__ alignas(16) __bf16 Wl[2][256 * LROW_];   // 2 x 20KB
    __shared__ alignas(16) __bf16 Al[2][128 * LROW_];   // 2 x 10KB

    const int tid  = threadIdx.x;
    const int lane = tid & 31;
    const int wave = tid >> 5;
    const int bw = wave & 3, ow = wave >> 2;
    const int n   = blockIdx.x;
    const int b0b = blockIdx.y * 128;
    const int o0b = blockIdx.z * 256;
    const int col = lane & 15;
    const int hi  = lane >> 4;
    const int alo = hi * 8;     // A-operand K shift per lane half

    // Cooperative async issue of K-step tile kt into LDS buffer buf.
    auto issue_tile = [&](int kt, int buf) {
        const int phase = (kt >= 2);
        const int K  = phase ? H_ : I_;
        const int ks = phase ? (kt - 2) * 32 : kt * 32;
        const __bf16* Wn = (phase ? Whhb : Wihb) + (size_t)n * G4_ * K;
        const __bf16* Ab = phase ? hb : xb;
        const unsigned ldsW = (unsigned)(size_t)&Wl[buf][0];
        const unsigned ldsA = (unsigned)(size_t)&Al[buf][0];
        #pragma unroll
        for (int i = 0; i < 4; ++i) {            // 256x32 weight tile
            const int c = tid + i * 256;
            const int row = c >> 2, cc = c & 3;
            async_ld16(ldsW + row * (LROW_ * 2) + cc * 16,
                       Wn + (size_t)(o0b + row) * K + ks + cc * 8);
        }
        #pragma unroll
        for (int i = 0; i < 2; ++i) {            // 128x32 activation tile
            const int c = tid + i * 256;
            const int row = c >> 2, cc = c & 3;
            async_ld16(ldsA + row * (LROW_ * 2) + cc * 16,
                       Ab + ((size_t)(b0b + row) * N_ + n) * K + ks + cc * 8);
        }
    };

    v8f acc[2][8];
    #pragma unroll
    for (int bt = 0; bt < 2; ++bt)
        #pragma unroll
        for (int ot = 0; ot < 8; ++ot)
            acc[bt][ot] = v8f{0.f, 0.f, 0.f, 0.f, 0.f, 0.f, 0.f, 0.f};

    auto compute = [&](int buf) {
        v16bf a[2];
        #pragma unroll
        for (int bt = 0; bt < 2; ++bt) {
            const __bf16* pa = &Al[buf][(bw * 32 + bt * 16 + col) * LROW_];
            const v8bf a0 = *reinterpret_cast<const v8bf*>(pa + alo);
            const v8bf a1 = *reinterpret_cast<const v8bf*>(pa + alo + 16);
            #pragma unroll
            for (int e = 0; e < 8; ++e) { a[bt][e] = a0[e]; a[bt][8 + e] = a1[e]; }
        }
        #pragma unroll
        for (int ot = 0; ot < 8; ++ot) {
            const __bf16* pw = &Wl[buf][(ow * 128 + ot * 16 + col) * LROW_];
            const v8bf w0 = *reinterpret_cast<const v8bf*>(pw + hi * 8);
            const v8bf w1 = *reinterpret_cast<const v8bf*>(pw + hi * 8 + 16);
            v16bf bb;
            #pragma unroll
            for (int e = 0; e < 8; ++e) { bb[e] = w0[e]; bb[8 + e] = w1[e]; }
            #pragma unroll
            for (int bt = 0; bt < 2; ++bt)
                acc[bt][ot] = __builtin_amdgcn_wmma_f32_16x16x32_bf16(
                    false, a[bt], false, bb, (short)0, acc[bt][ot],
                    false, false);
        }
    };

    // Software pipeline: prefetch tile kt+1 while computing tile kt.
    issue_tile(0, 0);
    #pragma unroll 1
    for (int kt = 0; kt < 9; ++kt) {
        issue_tile(kt + 1, (kt + 1) & 1);
        wait_async6();          // own tile-kt loads done (6 newest outstanding)
        __syncthreads();        // everyone's tile-kt loads visible
        compute(kt & 1);
        __syncthreads();        // finish reads before next overwrite
    }
    wait_async0();
    __syncthreads();
    compute(1);                 // kt = 9 -> buffer 1

    // Store pre-gates, layout [n][b][o]: 16 lanes x 2B = 32B contiguous runs.
    #pragma unroll
    for (int bt = 0; bt < 2; ++bt) {
        #pragma unroll
        for (int ot = 0; ot < 8; ++ot) {
            const int o = o0b + ow * 128 + ot * 16 + col;
            #pragma unroll
            for (int r = 0; r < 8; ++r) {
                const int brow = b0b + bw * 32 + bt * 16 + r + hi * 8;  // M=r+8*hi
                s[((size_t)n * B_ + brow) * G4_ + o] = (__bf16)acc[bt][ot][r];
            }
        }
    }
}

// ---------------------------------------------------------------------------
// Kernel 2: graph mixing (K=24 padded to one K=32 WMMA) + biases + LSTM.
// Grid: 2 blocks per batch element; block handles 8 h-tiles (one per wave).
// Cooperative coalesced load (256B runs) of the 24KB s-slice, transposed into
// a bank-skewed LDS tile [o_local][n]; pad columns n=24..31 zeroed (WMMA
// propagates NaN even against a zero A operand, and gx pads are zero).
// Per wave: 2 m-tiles x 4 gate-tiles = 8 WMMAs, then bias + LSTM elementwise.
// ---------------------------------------------------------------------------
__global__ __launch_bounds__(256)
void k_mix_lstm(const __bf16* __restrict__ s, const __bf16* __restrict__ gxb,
                const float* __restrict__ cx,
                const float* __restrict__ bih, const float* __restrict__ bhh,
                const int* __restrict__ tptr,
                float* __restrict__ hy, float* __restrict__ cy)
{
    __shared__ alignas(16) __bf16 Sl[512 * LROW_];      // 40KB: [o_local][npad]

    const int tid  = threadIdx.x;
    const int lane = tid & 31;
    const int wave = tid >> 5;
    const int b     = blockIdx.x >> 1;
    const int hhalf = blockIdx.x & 1;     // which 128-wide half of H
    const int col = lane & 15;
    const int hi  = lane >> 4;
    const int alo = hi * 8;

    // Zero pad columns n=24..31 (16B per row).
    v8bf zero8;
    #pragma unroll
    for (int e = 0; e < 8; ++e) zero8[e] = (__bf16)0.0f;
    #pragma unroll
    for (int i = 0; i < 2; ++i) {
        const int row = tid + i * 256;
        *reinterpret_cast<v8bf*>(&Sl[row * LROW_ + 24]) = zero8;
    }

    // Cooperative load + transpose: chunks c = n(24) x g(4) x q(16 x 16B),
    // covering o = g*256 + hhalf*128 + [0,128) for all 24 nodes.
    #pragma unroll
    for (int i = 0; i < 6; ++i) {
        const int c = tid + i * 256;                    // < 1536
        const int n = c >> 6;
        const int g = (c >> 4) & 3;
        const int q = c & 15;
        const v8bf v = *reinterpret_cast<const v8bf*>(
            s + ((size_t)n * B_ + b) * G4_ + g * H_ + hhalf * 128 + q * 8);
        const int rbase = (g * 128 + q * 8) * LROW_ + n;
        #pragma unroll
        for (int e = 0; e < 8; ++e) Sl[rbase + e * LROW_] = v[e];
    }
    __syncthreads();

    const int h = hhalf * 128 + wave * 16 + col;

    // B operand: per gate, rows = o_local, per-lane K(=n) = {8h..8h+7, 16+8h..16+8h+7}.
    v16bf Bt[4];
    #pragma unroll
    for (int g = 0; g < 4; ++g) {
        const __bf16* p = &Sl[(g * 128 + wave * 16 + col) * LROW_ + hi * 8];
        const v8bf c0 = *reinterpret_cast<const v8bf*>(p);
        const v8bf c1 = *reinterpret_cast<const v8bf*>(p + 16);
        #pragma unroll
        for (int e = 0; e < 8; ++e) { Bt[g][e] = c0[e]; Bt[g][8 + e] = c1[e]; }
    }

    // A operand: gx rows (zero-padded 32x32 bf16).
    v16bf At[2];
    #pragma unroll
    for (int mt = 0; mt < 2; ++mt) {
        const __bf16* gp = gxb + (mt * 16 + col) * 32 + alo;
        const v8bf g0 = *reinterpret_cast<const v8bf*>(gp);
        const v8bf g1 = *reinterpret_cast<const v8bf*>(gp + 16);
        #pragma unroll
        for (int e = 0; e < 8; ++e) { At[mt][e] = g0[e]; At[mt][8 + e] = g1[e]; }
    }

    v8f acc[2][4];
    #pragma unroll
    for (int mt = 0; mt < 2; ++mt)
        #pragma unroll
        for (int g = 0; g < 4; ++g)
            acc[mt][g] = __builtin_amdgcn_wmma_f32_16x16x32_bf16(
                false, At[mt], false, Bt[g], (short)0,
                v8f{0.f, 0.f, 0.f, 0.f, 0.f, 0.f, 0.f, 0.f}, false, false);

    float bsum[4];
    #pragma unroll
    for (int g = 0; g < 4; ++g)
        bsum[g] = bih[g * H_ + h] + bhh[g * H_ + h];

    const float phase = floorf((float)h * (8.0f / (float)(H_ - 1)) + 1.0f);
    const float tt = (float)(*tptr) + 1.0f;
    const float cmask = (fmodf(tt, phase) < 0.01f) ? 1.0f : 0.0f;

    #pragma unroll
    for (int mt = 0; mt < 2; ++mt) {
        const int mbase = mt * 16 + hi * 8;    // C/D: M = r + 8*hi
        #pragma unroll
        for (int r = 0; r < 8; ++r) {
            const int m = mbase + r;
            if (m < N_) {
                const size_t idx = ((size_t)b * N_ + m) * H_ + h;
                const float ig = sigmoidf(acc[mt][0][r] + bsum[0]);
                const float fg = sigmoidf(acc[mt][1][r] + bsum[1]);
                const float cg = tanhf(acc[mt][2][r] + bsum[2]);
                const float og = sigmoidf(acc[mt][3][r] + bsum[3]);
                const float c  = cx[idx];
                const float cv = cmask * (fg * c + ig * cg) + (1.0f - cmask) * c;
                cy[idx] = cv;
                hy[idx] = og * tanhf(cv);
            }
        }
    }
}

// ---------------------------------------------------------------------------
extern "C" void kernel_launch(void* const* d_in, const int* in_sizes, int n_in,
                              void* d_out, int out_size, void* d_ws, size_t ws_size,
                              hipStream_t stream)
{
    const float* x   = (const float*)d_in[0];
    const float* hx  = (const float*)d_in[1];
    const float* cx  = (const float*)d_in[2];
    const float* G   = (const float*)d_in[3];
    const float* Wih = (const float*)d_in[4];
    const float* Whh = (const float*)d_in[5];
    const float* bih = (const float*)d_in[6];
    const float* bhh = (const float*)d_in[7];
    const int*   t   = (const int*)d_in[8];

    float* hy     = (float*)d_out;
    float* cyp    = hy  + (size_t)B_ * N_ * H_;
    float* gx_out = cyp + (size_t)B_ * N_ * H_;

    const size_t nX   = (size_t)B_ * N_ * I_;     //  6,291,456
    const size_t nH   = (size_t)B_ * N_ * H_;     // 25,165,824
    const size_t nWih = (size_t)N_ * G4_ * I_;    //  1,572,864
    const size_t nWhh = (size_t)N_ * G4_ * H_;    //  6,291,456

    char* ws = (char*)d_ws;
    __bf16* gxb  = (__bf16*)ws;                          // 2 KB (32x32)
    size_t off = 4096;
    __bf16* xb   = (__bf16*)(ws + off); off += nX   * 2;
    __bf16* hb   = (__bf16*)(ws + off); off += nH   * 2;
    __bf16* Wihb = (__bf16*)(ws + off); off += nWih * 2;
    __bf16* Whhb = (__bf16*)(ws + off); off += nWhh * 2;
    __bf16* s    = (__bf16*)(ws + off);                  // ~201 MB, [n][b][o]

    k_gx<<<1, 256, 0, stream>>>(G, gxb, gx_out);
    k_tobf<<<dim3((unsigned)(nX   / 8 / 256)), 256, 0, stream>>>(x,   xb,   (int)(nX   / 8));
    k_tobf<<<dim3((unsigned)(nH   / 8 / 256)), 256, 0, stream>>>(hx,  hb,   (int)(nH   / 8));
    k_tobf<<<dim3((unsigned)(nWih / 8 / 256)), 256, 0, stream>>>(Wih, Wihb, (int)(nWih / 8));
    k_tobf<<<dim3((unsigned)(nWhh / 8 / 256)), 256, 0, stream>>>(Whh, Whhb, (int)(nWhh / 8));
    k_node_gemm<<<dim3(N_, B_ / 128, G4_ / 256), 256, 0, stream>>>(
        xb, hb, Wihb, Whhb, s);
    k_mix_lstm<<<dim3(B_ * 2), 256, 0, stream>>>(
        s, gxb, cx, bih, bhh, t, hy, cyp);
}
